// MLPRegularQuantised_53884659695675
// MI455X (gfx1250) — matrix operand order repaired and checked by reading, not compiled, not run on hardware
//
#include <hip/hip_runtime.h>

typedef __attribute__((ext_vector_type(16))) _Float16 v16h;
typedef __attribute__((ext_vector_type(8)))  _Float16 v8h;
typedef __attribute__((ext_vector_type(8)))  float    v8f;

#define WMMA_F16(a, b, c) \
  __builtin_amdgcn_wmma_f32_16x16x32_f16(false, (a), false, (b), (short)0, (c), false, false)

// QKeras quantized_bits(8,0,alpha=1,keep_negative=1): clip(round(x*128),-128,127)/128
__device__ __forceinline__ float qbits_q(float x) {
  float q = rintf(x * 128.0f);
  q = fminf(fmaxf(q, -128.0f), 127.0f);
  return q * (1.0f / 128.0f);
}
// QKeras quantized_relu(8,0): clip(round(x*256),0,255)/256
__device__ __forceinline__ float qrelu_q(float x) {
  float q = rintf(x * 256.0f);
  q = fminf(fmaxf(q, 0.0f), 255.0f);
  return q * (1.0f / 256.0f);
}

// Build one B fragment (16x16 output tile, K=32 slice) for this lane from
// row-major W[K][N]. CDNA5 16-bit B layout: lane L -> col n = L%16, group
// g = L/16; half h -> k = 32*kt + 8*g + (h&7) + 16*(h>>3). Out-of-range K/N padded with 0.
__device__ __forceinline__ v16h make_bfrag(const float* __restrict__ W, int K, int N,
                                           int kt, int nt, int lane, bool quant) {
  const int n = nt * 16 + (lane & 15);
  const int g = lane >> 4;
  v16h b;
#pragma unroll
  for (int h = 0; h < 16; ++h) {
    const int k = kt * 32 + g * 8 + (h & 7) + ((h >> 3) << 4);
    float v = 0.0f;
    if (k < K && n < N) {
      v = W[k * N + n];
      if (quant) v = qbits_q(v);
    }
    b[h] = (_Float16)v;
  }
  return b;
}

// A fragment from an LDS row (row-major [16][stride] f16). CDNA5 16-bit A
// layout: lane L -> row m = L%16, g = L/16; halves 0..7 = k kbase..kbase+7,
// halves 8..15 = k kbase+16..kbase+23, with kbase = 32*kt + 8*g.
// Two 16B LDS loads (ds_load_b128).
__device__ __forceinline__ v16h load_afrag(const _Float16* row, int kbase) {
  v8h lo = *(const v8h*)(row + kbase);
  v8h hi = *(const v8h*)(row + kbase + 16);
  return __builtin_shufflevector(lo, hi, 0, 1, 2, 3, 4, 5, 6, 7,
                                 8, 9, 10, 11, 12, 13, 14, 15);
}

// A fragment for layer-1 directly from global x (row of 48 f32), K padded to 64.
__device__ __forceinline__ v16h load_xfrag(const float* __restrict__ xrow, int g,
                                           int kt, bool valid) {
  float4 c0a = make_float4(0.f, 0.f, 0.f, 0.f), c0b = c0a, c1a = c0a, c1b = c0a;
  if (valid) {
    const float* p = xrow + 32 * kt + 8 * g;        // k in [0,48) always
    c0a = *(const float4*)(p);
    c0b = *(const float4*)(p + 4);
    if (kt == 0) {                                   // second chunk only in-range for kt==0
      c1a = *(const float4*)(p + 16);
      c1b = *(const float4*)(p + 20);
    }
  }
  v16h a;
  a[0]  = (_Float16)c0a.x; a[1]  = (_Float16)c0a.y; a[2]  = (_Float16)c0a.z; a[3]  = (_Float16)c0a.w;
  a[4]  = (_Float16)c0b.x; a[5]  = (_Float16)c0b.y; a[6]  = (_Float16)c0b.z; a[7]  = (_Float16)c0b.w;
  a[8]  = (_Float16)c1a.x; a[9]  = (_Float16)c1a.y; a[10] = (_Float16)c1a.z; a[11] = (_Float16)c1a.w;
  a[12] = (_Float16)c1b.x; a[13] = (_Float16)c1b.y; a[14] = (_Float16)c1b.z; a[15] = (_Float16)c1b.w;
  return a;
}

// Fused 4-layer quantized MLP. One wave = one 16-row tile per loop iteration.
__global__ void __launch_bounds__(256)
mlp_quant_fused_kernel(const float* __restrict__ x,
                       const float* __restrict__ w1, const float* __restrict__ b1,
                       const float* __restrict__ w2, const float* __restrict__ b2,
                       const float* __restrict__ w3, const float* __restrict__ b3,
                       const float* __restrict__ wo, const float* __restrict__ bo,
                       float* __restrict__ out, int B) {
  // Per-wave private activation staging (D layout -> A layout re-shuffle).
  // Row strides 72/40 halves: 16B-aligned rows, conflict-free bank quads.
  __shared__ __align__(16) _Float16 sh1[8][16 * 72];  // 16x64 h1 (padded)
  __shared__ __align__(16) _Float16 sh2[8][16 * 40];  // 16x32 h2 (padded)
  __shared__ __align__(16) _Float16 sh3[8][16 * 40];  // 16x32 h3 (padded)

  const int lane  = threadIdx.x & 31;
  const int wslot = threadIdx.x >> 5;
  const int g  = lane >> 4;
  const int nl = lane & 15;

  // ---------------- preamble: weight fragments resident in VGPRs ----------
  v16h bw1[2][4], bw2[2][2], bw3[2], bwo;
#pragma unroll
  for (int kt = 0; kt < 2; ++kt)
#pragma unroll
    for (int nt = 0; nt < 4; ++nt)
      bw1[kt][nt] = make_bfrag(w1, 48, 64, kt, nt, lane, true);
#pragma unroll
  for (int kt = 0; kt < 2; ++kt)
#pragma unroll
    for (int nt = 0; nt < 2; ++nt)
      bw2[kt][nt] = make_bfrag(w2, 64, 32, kt, nt, lane, true);
#pragma unroll
  for (int nt = 0; nt < 2; ++nt)
    bw3[nt] = make_bfrag(w3, 32, 32, 0, nt, lane, true);
  bwo = make_bfrag(wo, 32, 5, 0, 0, lane, false);  // float head: no fake-quant

  float bq1[4], bq2[2], bq3[2];
#pragma unroll
  for (int nt = 0; nt < 4; ++nt) bq1[nt] = qbits_q(b1[nt * 16 + nl]);
#pragma unroll
  for (int nt = 0; nt < 2; ++nt) bq2[nt] = qbits_q(b2[nt * 16 + nl]);
#pragma unroll
  for (int nt = 0; nt < 2; ++nt) bq3[nt] = qbits_q(b3[nt * 16 + nl]);
  const float bqo = (nl < 5) ? bo[nl] : 0.0f;

  _Float16* h1 = sh1[wslot];
  _Float16* h2 = sh2[wslot];
  _Float16* h3 = sh3[wslot];

  const int tiles  = (B + 15) >> 4;
  const int waveId = blockIdx.x * (blockDim.x >> 5) + wslot;
  const int nWaves = gridDim.x * (blockDim.x >> 5);

  for (int t = waveId; t < tiles; t += nWaves) {
    const int  row   = t * 16 + nl;
    const bool valid = row < B;
    const float* xrow = x + (size_t)row * 48;

    // Prefetch next tile's rows into cache (global_prefetch_b8).
    {
      const long prow = (long)(t + nWaves) * 16 + nl;
      if (prow < (long)B) __builtin_prefetch(x + prow * 48, 0, 0);
    }

    // ---- layer 1: [16x48] @ qbits(w1)[48x64] + qbits(b1), qrelu ----
    v16h a0 = load_xfrag(xrow, g, 0, valid);
    v16h a1 = load_xfrag(xrow, g, 1, valid);
#pragma unroll
    for (int nt = 0; nt < 4; ++nt) {
      v8f c;
#pragma unroll
      for (int r = 0; r < 8; ++r) c[r] = bq1[nt];
      c = WMMA_F16(a0, bw1[0][nt], c);
      c = WMMA_F16(a1, bw1[1][nt], c);
#pragma unroll
      for (int r = 0; r < 8; ++r) {
        const int m = r + 8 * g;                       // D layout: lane nl, VGPR r
        h1[m * 72 + nt * 16 + nl] = (_Float16)qrelu_q(c[r]);
      }
    }

    // ---- layer 2: [16x64] @ qbits(w2)[64x32] + qbits(b2), qrelu ----
    {
      const _Float16* arow = h1 + nl * 72;
      v16h a20 = load_afrag(arow, 8 * g);
      v16h a21 = load_afrag(arow, 32 + 8 * g);
#pragma unroll
      for (int nt = 0; nt < 2; ++nt) {
        v8f c;
#pragma unroll
        for (int r = 0; r < 8; ++r) c[r] = bq2[nt];
        c = WMMA_F16(a20, bw2[0][nt], c);
        c = WMMA_F16(a21, bw2[1][nt], c);
#pragma unroll
        for (int r = 0; r < 8; ++r)
          h2[(r + 8 * g) * 40 + nt * 16 + nl] = (_Float16)qrelu_q(c[r]);
      }
    }

    // ---- layer 3: [16x32] @ qbits(w3)[32x32] + qbits(b3), qrelu ----
    {
      v16h a3 = load_afrag(h2 + nl * 40, 8 * g);
#pragma unroll
      for (int nt = 0; nt < 2; ++nt) {
        v8f c;
#pragma unroll
        for (int r = 0; r < 8; ++r) c[r] = bq3[nt];
        c = WMMA_F16(a3, bw3[nt], c);
#pragma unroll
        for (int r = 0; r < 8; ++r)
          h3[(r + 8 * g) * 40 + nt * 16 + nl] = (_Float16)qrelu_q(c[r]);
      }
    }

    // ---- head: [16x32] @ wout[32x5] + bout (float, no quant/relu) ----
    {
      v16h ao = load_afrag(h3 + nl * 40, 8 * g);
      v8f c;
#pragma unroll
      for (int r = 0; r < 8; ++r) c[r] = bqo;
      c = WMMA_F16(ao, bwo, c);
      if (nl < 5) {
#pragma unroll
        for (int r = 0; r < 8; ++r) {
          const int orow = t * 16 + (r + 8 * g);
          if (orow < B) out[(size_t)orow * 5 + nl] = c[r];
        }
      }
    }
  }
}

extern "C" void kernel_launch(void* const* d_in, const int* in_sizes, int n_in,
                              void* d_out, int out_size, void* d_ws, size_t ws_size,
                              hipStream_t stream) {
  const float* x  = (const float*)d_in[0];
  const float* w1 = (const float*)d_in[1];
  const float* b1 = (const float*)d_in[2];
  const float* w2 = (const float*)d_in[3];
  const float* b2 = (const float*)d_in[4];
  const float* w3 = (const float*)d_in[5];
  const float* b3 = (const float*)d_in[6];
  const float* wo = (const float*)d_in[7];
  const float* bo = (const float*)d_in[8];
  float* out = (float*)d_out;

  const int B     = in_sizes[0] / 48;     // x is [B,16,3]
  const int tiles = (B + 15) / 16;
  int blocks = (tiles + 7) / 8;           // 8 waves per block, 1 tile/wave/iter
  if (blocks > 1024) blocks = 1024;       // grid-stride beyond this
  if (blocks < 1) blocks = 1;

  mlp_quant_fused_kernel<<<blocks, 256, 0, stream>>>(x, w1, b1, w2, b2, w3, b3,
                                                     wo, bo, out, B);
}
